// MultiHeadAttention_2130303779051
// MI455X (gfx1250) — matrix operand order repaired and checked by reading, not compiled, not run on hardware
//
#include <hip/hip_runtime.h>

// Problem constants (match reference)
#define B_  4
#define L_  2048
#define D_  1024
#define H_  16
#define DK_ 64
#define DV_ 64
#define N_  (B_ * L_)          // 8192 rows
#define BH_ (B_ * H_)          // 64

typedef __attribute__((ext_vector_type(16))) __bf16 v16bf;
typedef __attribute__((ext_vector_type(8)))  __bf16 v8bf;
typedef __attribute__((ext_vector_type(4)))  __bf16 v4bf;
typedef __attribute__((ext_vector_type(8)))  float  v8f;

union VAB { v16bf v; v8bf h[2]; };

// ---- WMMA fragment loaders (wave32 layouts per CDNA5 ISA 7.12.2) ----

// A-matrix 16x32 bf16: lane row = lane&15; low lanes K {0..7,16..23}, high lanes +8.
// Caller passes p0 = rowptr + kbase + ((lane>>4)*8); runs at p0 and p0+16.
__device__ __forceinline__ v16bf load_a_bf16(const __bf16* p0) {
    VAB a;
    a.h[0] = *(const v8bf*)(p0);
    a.h[1] = *(const v8bf*)(p0 + 16);
    return a.v;
}

// Same fragment but source is fp32 (attn matrix); native cvt to bf16.
__device__ __forceinline__ v16bf load_a_f32(const float* p0) {
    float4 f0 = *(const float4*)(p0);
    float4 f1 = *(const float4*)(p0 + 4);
    float4 f2 = *(const float4*)(p0 + 16);
    float4 f3 = *(const float4*)(p0 + 20);
    v16bf a;
    a[0]  = (__bf16)f0.x; a[1]  = (__bf16)f0.y; a[2]  = (__bf16)f0.z; a[3]  = (__bf16)f0.w;
    a[4]  = (__bf16)f1.x; a[5]  = (__bf16)f1.y; a[6]  = (__bf16)f1.z; a[7]  = (__bf16)f1.w;
    a[8]  = (__bf16)f2.x; a[9]  = (__bf16)f2.y; a[10] = (__bf16)f2.z; a[11] = (__bf16)f2.w;
    a[12] = (__bf16)f3.x; a[13] = (__bf16)f3.y; a[14] = (__bf16)f3.z; a[15] = (__bf16)f3.w;
    return a;
}

// B-matrix 32x16 bf16: lane col = lane&15; low lanes K {0..7,16..23}, high lanes +8.
// Caller passes p = colptr + kbase + ((lane>>4)*8); runs at p and p+16.
__device__ __forceinline__ v16bf load_b_bf16(const __bf16* p) {
    VAB b;
    b.h[0] = *(const v8bf*)(p);
    b.h[1] = *(const v8bf*)(p + 16);
    return b.v;
}

__device__ __forceinline__ v8f wmma_bf16(v16bf a, v16bf b, v8f c) {
    // (neg_a, A, neg_b, B, c_mod, C, reuse_a, reuse_b)
    return __builtin_amdgcn_wmma_f32_16x16x32_bf16(false, a, false, b, (short)0, c,
                                                   false, false);
}

// ---------------- Kernels ----------------

// fp32 -> bf16 bulk convert, 4 elements/thread (float4 in, 8B out)
__global__ void cvt4_kernel(const float* __restrict__ src,
                            __bf16* __restrict__ dst, int n4) {
    int i = blockIdx.x * blockDim.x + threadIdx.x;
    if (i < n4) {
        float4 f = ((const float4*)src)[i];
        v4bf o;
        o[0] = (__bf16)f.x; o[1] = (__bf16)f.y; o[2] = (__bf16)f.z; o[3] = (__bf16)f.w;
        ((v4bf*)dst)[i] = o;
    }
}

// C[N_,1024] = X @ W^T ; each wave computes a 16x64 strip (4 accumulators, A reused 4x).
// vmode==0: dst[((b*H+h)*L + l)*DK + d]   (Q,K)
// vmode==1: dst[((b*H+h)*DV + d)*L + l]   (V, transposed so attn@V B-frags are contiguous)
__global__ void proj_kernel(const __bf16* __restrict__ X,   // [N_, D_] bf16
                            const __bf16* __restrict__ Wb,  // [1024, D_] bf16
                            __bf16* __restrict__ dst, int vmode) {
    const int lane = threadIdx.x & 31;
    const int wave = threadIdx.x >> 5;
    const int wi   = blockIdx.x * 4 + wave;    // 0..15 : which 64-col strip
    const int row0 = blockIdx.y * 16;          // 0..511 row tiles
    const int cb   = wi * 64;
    const int off8  = (lane >> 4) * 8;
    const int off16 = (lane >> 4) * 16;
    const int ml = lane & 15;

    const __bf16* arow = X + (long)(row0 + ml) * D_;
    const __bf16* b0 = Wb + (long)(cb +  0 + ml) * D_;
    const __bf16* b1 = Wb + (long)(cb + 16 + ml) * D_;
    const __bf16* b2 = Wb + (long)(cb + 32 + ml) * D_;
    const __bf16* b3 = Wb + (long)(cb + 48 + ml) * D_;

    v8f acc0 = {}, acc1 = {}, acc2 = {}, acc3 = {};
    #pragma unroll 4
    for (int kb = 0; kb < D_; kb += 32) {
        v16bf a = load_a_bf16(arow + kb + off8);
        acc0 = wmma_bf16(a, load_b_bf16(b0 + kb + off8), acc0);
        acc1 = wmma_bf16(a, load_b_bf16(b1 + kb + off8), acc1);
        acc2 = wmma_bf16(a, load_b_bf16(b2 + kb + off8), acc2);
        acc3 = wmma_bf16(a, load_b_bf16(b3 + kb + off8), acc3);
    }

    #pragma unroll
    for (int j = 0; j < 4; j++) {
        const v8f* acc = (j == 0) ? &acc0 : (j == 1) ? &acc1 : (j == 2) ? &acc2 : &acc3;
        const int n = cb + j * 16 + ml;
        const int h = n / DK_, d = n % DK_;
        #pragma unroll
        for (int r = 0; r < 8; r++) {
            int m  = row0 + r + ((lane >> 4) << 3);
            int bb = m / L_, l = m % L_;
            long idx = (vmode == 0)
                ? ((long)(bb * H_ + h) * L_ + l) * DK_ + d
                : ((long)(bb * H_ + h) * DV_ + d) * (long)L_ + l;
            dst[idx] = (__bf16)((*acc)[r]);
        }
    }
}

// scores[bh, q, k] = (Qh . Kh) / sqrt(DK); each wave does a 16(q) x 64(k) strip.
__global__ void scores_kernel(const __bf16* __restrict__ Qh,   // [BH, L, DK]
                              const __bf16* __restrict__ Kh,   // [BH, L, DK]
                              float* __restrict__ attn) {      // [BH, L, L]
    const int lane = threadIdx.x & 31;
    const int wave = threadIdx.x >> 5;
    const int kg   = blockIdx.x * 4 + wave;    // 0..31 : 64-wide k strip
    const int q0   = blockIdx.y * 16;          // 0..127 q tiles
    const int bh   = blockIdx.z;               // 0..63
    const int k0   = kg * 64;
    const int off8  = (lane >> 4) * 8;
    const int off16 = (lane >> 4) * 16;
    const int ml = lane & 15;

    const __bf16* qrow = Qh + ((long)bh * L_ + q0 + ml) * DK_;
    const __bf16* kr0 = Kh + ((long)bh * L_ + k0 +  0 + ml) * DK_;
    const __bf16* kr1 = Kh + ((long)bh * L_ + k0 + 16 + ml) * DK_;
    const __bf16* kr2 = Kh + ((long)bh * L_ + k0 + 32 + ml) * DK_;
    const __bf16* kr3 = Kh + ((long)bh * L_ + k0 + 48 + ml) * DK_;

    v8f acc0 = {}, acc1 = {}, acc2 = {}, acc3 = {};
    #pragma unroll
    for (int kb = 0; kb < DK_; kb += 32) {
        v16bf a = load_a_bf16(qrow + kb + off8);
        acc0 = wmma_bf16(a, load_b_bf16(kr0 + kb + off8), acc0);
        acc1 = wmma_bf16(a, load_b_bf16(kr1 + kb + off8), acc1);
        acc2 = wmma_bf16(a, load_b_bf16(kr2 + kb + off8), acc2);
        acc3 = wmma_bf16(a, load_b_bf16(kr3 + kb + off8), acc3);
    }

    const float scale = 0.125f;                // 1/sqrt(64)
    float* outp = attn + (long)bh * L_ * L_;
    #pragma unroll
    for (int j = 0; j < 4; j++) {
        const v8f* acc = (j == 0) ? &acc0 : (j == 1) ? &acc1 : (j == 2) ? &acc2 : &acc3;
        const int n = k0 + j * 16 + ml;
        #pragma unroll
        for (int r = 0; r < 8; r++) {
            int m = q0 + r + ((lane >> 4) << 3);
            outp[(long)m * L_ + n] = (*acc)[r] * scale;
        }
    }
}

// in-place row softmax over L_=2048, one 256-thread block per row
__global__ void softmax_kernel(float* __restrict__ attn) {
    __shared__ float red[256];
    float* p = attn + (long)blockIdx.x * L_;
    const int t = threadIdx.x;
    float vals[8];
    float mx = -3.4e38f;
    #pragma unroll
    for (int i = 0; i < 8; i++) { vals[i] = p[t + i * 256]; mx = fmaxf(mx, vals[i]); }
    red[t] = mx; __syncthreads();
    for (int s = 128; s > 0; s >>= 1) { if (t < s) red[t] = fmaxf(red[t], red[t + s]); __syncthreads(); }
    mx = red[0]; __syncthreads();
    float sum = 0.f;
    #pragma unroll
    for (int i = 0; i < 8; i++) { vals[i] = __expf(vals[i] - mx); sum += vals[i]; }
    red[t] = sum; __syncthreads();
    for (int s = 128; s > 0; s >>= 1) { if (t < s) red[t] += red[t + s]; __syncthreads(); }
    const float inv = 1.0f / red[0];
    #pragma unroll
    for (int i = 0; i < 8; i++) p[t + i * 256] = vals[i] * inv;
}

// ctx[b*L+q, h*DV+d] = sum_k attn[bh,q,k] * V[bh,k,d]; V stored transposed [BH, DV, L].
// Each wave does a 16(q) x 64(all DV) strip: the fp32->bf16 convert of the attn
// fragment is amortized over 4 WMMAs.
__global__ void ctx_kernel(const float*  __restrict__ attn,  // [BH, L, L] fp32
                           const __bf16* __restrict__ Vt,    // [BH, DV, L] bf16
                           __bf16* __restrict__ ctx) {       // [N_, H_*DV_] bf16
    const int lane = threadIdx.x & 31;
    const int wave = threadIdx.x >> 5;
    const int tileQ = blockIdx.x * 4 + wave;  // 0..127
    const int bh    = blockIdx.y;             // 0..63
    const int q0 = tileQ * 16;
    const int off8  = (lane >> 4) * 8;
    const int off16 = (lane >> 4) * 16;
    const int ml = lane & 15;

    const float*  arow = attn + ((long)bh * L_ + q0 + ml) * (long)L_;
    const __bf16* v0 = Vt + ((long)bh * DV_ +  0 + ml) * (long)L_;
    const __bf16* v1 = Vt + ((long)bh * DV_ + 16 + ml) * (long)L_;
    const __bf16* v2 = Vt + ((long)bh * DV_ + 32 + ml) * (long)L_;
    const __bf16* v3 = Vt + ((long)bh * DV_ + 48 + ml) * (long)L_;

    v8f acc0 = {}, acc1 = {}, acc2 = {}, acc3 = {};
    #pragma unroll 2
    for (int kb = 0; kb < L_; kb += 32) {
        __builtin_prefetch(arow + kb + 128, 0, 1);   // global_prefetch_b8
        v16bf a = load_a_f32(arow + kb + off8);
        acc0 = wmma_bf16(a, load_b_bf16(v0 + kb + off8), acc0);
        acc1 = wmma_bf16(a, load_b_bf16(v1 + kb + off8), acc1);
        acc2 = wmma_bf16(a, load_b_bf16(v2 + kb + off8), acc2);
        acc3 = wmma_bf16(a, load_b_bf16(v3 + kb + off8), acc3);
    }

    const int h = bh % H_, bb = bh / H_;
    #pragma unroll
    for (int j = 0; j < 4; j++) {
        const v8f* acc = (j == 0) ? &acc0 : (j == 1) ? &acc1 : (j == 2) ? &acc2 : &acc3;
        const int n = j * 16 + ml;
        #pragma unroll
        for (int r = 0; r < 8; r++) {
            int m = q0 + r + ((lane >> 4) << 3);  // query index within L
            long idx = ((long)(bb * L_ + m)) * (H_ * DV_) + h * DV_ + n;
            ctx[idx] = (__bf16)((*acc)[r]);
        }
    }
}

// out[N_,D_] = ctx @ Wfc^T + residual(q); 16x64 strip per wave.
__global__ void fc_kernel(const __bf16* __restrict__ ctx,  // [N_, 1024] bf16
                          const __bf16* __restrict__ Wf,   // [1024, 1024] bf16
                          const float*  __restrict__ resid,// q [N_, D_]
                          float* __restrict__ out) {       // [N_, D_]
    const int lane = threadIdx.x & 31;
    const int wave = threadIdx.x >> 5;
    const int wi   = blockIdx.x * 4 + wave;   // 0..15
    const int row0 = blockIdx.y * 16;         // 0..511
    const int cb   = wi * 64;
    const int off8  = (lane >> 4) * 8;
    const int off16 = (lane >> 4) * 16;
    const int ml = lane & 15;

    const __bf16* arow = ctx + (long)(row0 + ml) * (H_ * DV_);
    const __bf16* b0 = Wf + (long)(cb +  0 + ml) * (H_ * DV_);
    const __bf16* b1 = Wf + (long)(cb + 16 + ml) * (H_ * DV_);
    const __bf16* b2 = Wf + (long)(cb + 32 + ml) * (H_ * DV_);
    const __bf16* b3 = Wf + (long)(cb + 48 + ml) * (H_ * DV_);

    v8f acc0 = {}, acc1 = {}, acc2 = {}, acc3 = {};
    #pragma unroll 4
    for (int kb = 0; kb < H_ * DV_; kb += 32) {
        v16bf a = load_a_bf16(arow + kb + off8);
        acc0 = wmma_bf16(a, load_b_bf16(b0 + kb + off8), acc0);
        acc1 = wmma_bf16(a, load_b_bf16(b1 + kb + off8), acc1);
        acc2 = wmma_bf16(a, load_b_bf16(b2 + kb + off8), acc2);
        acc3 = wmma_bf16(a, load_b_bf16(b3 + kb + off8), acc3);
    }

    #pragma unroll
    for (int j = 0; j < 4; j++) {
        const v8f* acc = (j == 0) ? &acc0 : (j == 1) ? &acc1 : (j == 2) ? &acc2 : &acc3;
        const int n = cb + j * 16 + ml;
        #pragma unroll
        for (int r = 0; r < 8; r++) {
            int m = row0 + r + ((lane >> 4) << 3);
            long idx = (long)m * D_ + n;
            out[idx] = (*acc)[r] + resid[idx];
        }
    }
}

// in-place LayerNorm over D_=1024, one 256-thread block per row
__global__ void ln_kernel(float* __restrict__ out,
                          const float* __restrict__ g,
                          const float* __restrict__ bta) {
    __shared__ float red[256];
    float* p = out + (long)blockIdx.x * D_;
    const int t = threadIdx.x;
    float v[4], s = 0.f;
    #pragma unroll
    for (int i = 0; i < 4; i++) { v[i] = p[t + i * 256]; s += v[i]; }
    red[t] = s; __syncthreads();
    for (int st = 128; st > 0; st >>= 1) { if (t < st) red[t] += red[t + st]; __syncthreads(); }
    const float mu = red[0] * (1.0f / D_); __syncthreads();
    float s2 = 0.f;
    #pragma unroll
    for (int i = 0; i < 4; i++) { float d = v[i] - mu; s2 += d * d; }
    red[t] = s2; __syncthreads();
    for (int st = 128; st > 0; st >>= 1) { if (t < st) red[t] += red[t + st]; __syncthreads(); }
    const float inv = rsqrtf(red[0] * (1.0f / D_) + 1e-6f);
    #pragma unroll
    for (int i = 0; i < 4; i++) {
        int c = t + i * 256;
        p[c] = (v[i] - mu) * inv * g[c] + bta[c];
    }
}

// ---------------- Host launcher ----------------

extern "C" void kernel_launch(void* const* d_in, const int* in_sizes, int n_in,
                              void* d_out, int out_size, void* d_ws, size_t ws_size,
                              hipStream_t stream) {
    (void)in_sizes; (void)n_in; (void)out_size; (void)ws_size;
    const float* q    = (const float*)d_in[0];
    const float* k    = (const float*)d_in[1];
    const float* v    = (const float*)d_in[2];
    const float* Wq   = (const float*)d_in[3];
    const float* Wk   = (const float*)d_in[4];
    const float* Wv   = (const float*)d_in[5];
    const float* Wfc  = (const float*)d_in[6];
    const float* ln_g = (const float*)d_in[7];
    const float* ln_b = (const float*)d_in[8];

    float* out  = (float*)d_out;                 // [N_, D_]  (output 0)
    float* attn = out + (long)N_ * D_;           // [BH, L, L] (output 1)

    // Workspace carve-up (~123 MB total)
    __bf16* Wq_b = (__bf16*)d_ws;                        // 1M elems each
    __bf16* Wk_b = Wq_b + 1024 * 1024;
    __bf16* Wv_b = Wk_b + 1024 * 1024;
    __bf16* Wf_b = Wv_b + 1024 * 1024;
    __bf16* Xq_b = Wf_b + 1024 * 1024;                   // [N_, D_] each, 8.39M elems
    __bf16* Xk_b = Xq_b + (long)N_ * D_;
    __bf16* Xv_b = Xk_b + (long)N_ * D_;
    __bf16* Qh   = Xv_b + (long)N_ * D_;                 // [BH,L,DK]
    __bf16* Kh   = Qh + (long)BH_ * L_ * DK_;
    __bf16* Vt   = Kh + (long)BH_ * L_ * DK_;            // [BH,DV,L]
    __bf16* ctx  = Vt + (long)BH_ * L_ * DV_;            // [N_, H*DV]

    const int WN4 = (1024 * 1024) / 4;
    const int XN4 = (N_ * D_) / 4;
    cvt4_kernel<<<WN4 / 256, 256, 0, stream>>>(Wq,  Wq_b, WN4);
    cvt4_kernel<<<WN4 / 256, 256, 0, stream>>>(Wk,  Wk_b, WN4);
    cvt4_kernel<<<WN4 / 256, 256, 0, stream>>>(Wv,  Wv_b, WN4);
    cvt4_kernel<<<WN4 / 256, 256, 0, stream>>>(Wfc, Wf_b, WN4);
    cvt4_kernel<<<XN4 / 256, 256, 0, stream>>>(q, Xq_b, XN4);
    cvt4_kernel<<<XN4 / 256, 256, 0, stream>>>(k, Xk_b, XN4);
    cvt4_kernel<<<XN4 / 256, 256, 0, stream>>>(v, Xv_b, XN4);

    proj_kernel<<<dim3(4, 512), 128, 0, stream>>>(Xq_b, Wq_b, Qh, 0);
    proj_kernel<<<dim3(4, 512), 128, 0, stream>>>(Xk_b, Wk_b, Kh, 0);
    proj_kernel<<<dim3(4, 512), 128, 0, stream>>>(Xv_b, Wv_b, Vt, 1);

    scores_kernel<<<dim3(8, 128, 64), 128, 0, stream>>>(Qh, Kh, attn);
    softmax_kernel<<<BH_ * L_, 256, 0, stream>>>(attn);
    ctx_kernel<<<dim3(32, 64), 128, 0, stream>>>(attn, Vt, ctx);
    fc_kernel<<<dim3(4, 512), 128, 0, stream>>>(ctx, Wf_b, q, out);
    ln_kernel<<<N_, 256, 0, stream>>>(out, ln_g, ln_b);
}
